// MultiheadAttention_79791902425338
// MI455X (gfx1250) — matrix operand order repaired and checked by reading, not compiled, not run on hardware
//
#include <hip/hip_runtime.h>
#include <hip/hip_bf16.h>

// ---------------------------------------------------------------------------
// MHA forward for MI455X (gfx1250, wave32, WMMA f32_16x16x32_f16)
// B=4 S=2048 D=1024 H=16 DK=64
// ---------------------------------------------------------------------------

typedef _Float16 half_t;
typedef __attribute__((ext_vector_type(16))) _Float16 v16h;
typedef __attribute__((ext_vector_type(8)))  float    v8f;
typedef int v4i_g __attribute__((vector_size(4 * sizeof(int))));

#define WMMA_F16(a, b, c) \
  __builtin_amdgcn_wmma_f32_16x16x32_f16(false, (a), false, (b), (short)0, (c), false, false)

#ifndef __has_builtin
#define __has_builtin(x) 0
#endif
#if __has_builtin(__builtin_amdgcn_global_load_async_to_lds_b128)
#define HAS_ASYNC_BUILTIN 1
#else
#define HAS_ASYNC_BUILTIN 0
#endif
#if __has_builtin(__builtin_amdgcn_s_wait_asynccnt)
#define HAS_WAITA_BUILTIN 1
#else
#define HAS_WAITA_BUILTIN 0
#endif

static constexpr int Bc = 4, Sc = 2048, Dc = 1024, Hc = 16, DKc = 64;
static constexpr int GM = Bc * Sc;   // 8192
static constexpr int GN = Dc;        // 1024
static constexpr int GK = Dc;        // 1024
static constexpr int BM = 128, BN = 128, BK = 32;
static constexpr int LDT = BK + 8;   // LDS row stride in halves (80 B, 16B-aligned)

union Frag16 { v16h v; uint4 u[2]; };

// ---- CDNA5 async global->LDS copy (ASYNCcnt), 16 bytes per lane ------------
__device__ __forceinline__ void async_b128(half_t* lds_dst, const half_t* gsrc) {
#if HAS_ASYNC_BUILTIN
  __builtin_amdgcn_global_load_async_to_lds_b128(
      (__attribute__((address_space(1))) v4i_g*)(uintptr_t)(const void*)gsrc,
      (__attribute__((address_space(3))) v4i_g*)(uint32_t)(uintptr_t)(void*)lds_dst,
      0, 0);
#elif defined(__AMDGCN__)
  asm volatile("global_load_async_to_lds_b128 %0, %1, off"
               :: "v"((uint32_t)(uintptr_t)(void*)lds_dst), "v"(gsrc)
               : "memory");
#endif
}

template <int N>
__device__ __forceinline__ void wait_async() {
#if HAS_WAITA_BUILTIN
  __builtin_amdgcn_s_wait_asynccnt(N);
#elif defined(__AMDGCN__)
  asm volatile("s_wait_asynccnt %0" :: "i"(N) : "memory");
#endif
}

// Load WMMA A-fragment (16x32 f16): lane holds row M=lane&15; K chunks at
// {koff..koff+7} and {16+koff..16+koff+7} with koff = (lane>=16)?8:0.
__device__ __forceinline__ v16h load_a_frag(const half_t* rowbase, int koff) {
  Frag16 f;
  f.u[0] = *(const uint4*)(rowbase + koff);
  f.u[1] = *(const uint4*)(rowbase + 16 + koff);
  return f.v;
}

// Load WMMA B-fragment (32x16 f16): lane holds col N=lane&15; K chunks at
// {koff..koff+7} and {16+koff..16+koff+7} with koff = (lane>=16)?8:0.
__device__ __forceinline__ v16h load_b_frag(const half_t* p) {
  Frag16 f;
  f.u[0] = *(const uint4*)(p);
  f.u[1] = *(const uint4*)(p + 16);
  return f.v;
}

// ---------------------------------------------------------------------------
// Projection GEMM:  Y[m,n] = sum_k A[m,k] * W[n,k] + bias[n]
// MODE 0: A=f32, store f16 at [B,H,S,DK]   (Q, K heads)
// MODE 1: A=f32, store f16 at [B,H,DK,S]   (V heads, transposed for P@V)
// MODE 2: A=f16, store f32 at [M,N]        (output projection)
// Block: 256 thr (8 waves). Block tile 128x128, wave tile 64x32, K-step 32.
// ---------------------------------------------------------------------------
template <int MODE>
__global__ __launch_bounds__(256) void proj_gemm(const void* __restrict__ Ain,
                                                 const float* __restrict__ W,
                                                 const float* __restrict__ bias,
                                                 void* __restrict__ Out) {
  __shared__ __align__(16) half_t As[BM * LDT];
  __shared__ __align__(16) half_t Bs[BN * LDT];

  const int tid  = threadIdx.x;
  const int lane = tid & 31;
  const int wave = tid >> 5;
  const int l15  = lane & 15;
  const int ah   = lane >> 4;                 // half-wave selector
  const int wm   = (wave >> 2) * 64;          // wave M offset within block tile
  const int wn   = (wave & 3) * 32;           // wave N offset within block tile
  const int mblk = blockIdx.y * BM;
  const int nblk = blockIdx.x * BN;

  const int lr = tid >> 1;                    // 0..127 tile row loaded by thread
  const int lc = (tid & 1) * 16;              // 0 or 16 within 32-wide K slab

  v8f acc[4][2];
#pragma unroll
  for (int mi = 0; mi < 4; mi++)
#pragma unroll
    for (int ni = 0; ni < 2; ni++)
#pragma unroll
      for (int v = 0; v < 8; v++) acc[mi][ni][v] = 0.0f;

  for (int kb = 0; kb < GK; kb += BK) {
    // ---- stage A tile (with fp32 -> f16 conversion unless MODE==2) ----
    if (MODE == 2) {
      const half_t* src = (const half_t*)Ain + (size_t)(mblk + lr) * GK + kb + lc;
      *(uint4*)&As[lr * LDT + lc]     = *(const uint4*)(src);
      *(uint4*)&As[lr * LDT + lc + 8] = *(const uint4*)(src + 8);
      if (kb + BK < GK) __builtin_prefetch(src + BK, 0, 1);
    } else {
      const float* src = (const float*)Ain + (size_t)(mblk + lr) * GK + kb + lc;
      float4 f[4];
#pragma unroll
      for (int j = 0; j < 4; j++) f[j] = ((const float4*)src)[j];
      union { half_t h[16]; uint4 u[2]; } t;
      const float* ff = (const float*)f;
#pragma unroll
      for (int j = 0; j < 16; j++) t.h[j] = (half_t)ff[j];
      *(uint4*)&As[lr * LDT + lc]     = t.u[0];
      *(uint4*)&As[lr * LDT + lc + 8] = t.u[1];
      if (kb + BK < GK) __builtin_prefetch(src + BK, 0, 1);
    }
    // ---- stage B tile: W[n, k] (row n of W, contiguous in k) ----
    {
      const float* src = W + (size_t)(nblk + lr) * GK + kb + lc;
      float4 f[4];
#pragma unroll
      for (int j = 0; j < 4; j++) f[j] = ((const float4*)src)[j];
      union { half_t h[16]; uint4 u[2]; } t;
      const float* ff = (const float*)f;
#pragma unroll
      for (int j = 0; j < 16; j++) t.h[j] = (half_t)ff[j];
      *(uint4*)&Bs[lr * LDT + lc]     = t.u[0];
      *(uint4*)&Bs[lr * LDT + lc + 8] = t.u[1];
      if (kb + BK < GK) __builtin_prefetch(src + BK, 0, 1);
    }
    __syncthreads();

    // ---- wave-level WMMAs: 4 (M) x 2 (N) tiles of 16x16, K=32 ----
    v16h afr[4];
#pragma unroll
    for (int mi = 0; mi < 4; mi++)
      afr[mi] = load_a_frag(&As[(wm + mi * 16 + l15) * LDT], ah * 8);
    v16h bfr[2];
#pragma unroll
    for (int ni = 0; ni < 2; ni++)
      bfr[ni] = load_b_frag(&Bs[(wn + ni * 16 + l15) * LDT + ah * 8]);
#pragma unroll
    for (int mi = 0; mi < 4; mi++)
#pragma unroll
      for (int ni = 0; ni < 2; ni++)
        acc[mi][ni] = WMMA_F16(afr[mi], bfr[ni], acc[mi][ni]);

    __syncthreads();
  }

  // ---- epilogue: C layout = row (VGPR + 8*half), col (lane&15) ----
#pragma unroll
  for (int ni = 0; ni < 2; ni++) {
    const int gn = nblk + wn + ni * 16 + l15;
    const float bv = bias[gn];
    const int hh = gn >> 6;        // head index
    const int dk = gn & 63;        // dim within head
#pragma unroll
    for (int mi = 0; mi < 4; mi++) {
      const int m0 = mblk + wm + mi * 16 + ah * 8;
#pragma unroll
      for (int v = 0; v < 8; v++) {
        const int gm = m0 + v;
        const float val = acc[mi][ni][v] + bv;
        const int bb = gm >> 11;   // / S
        const int ss = gm & 2047;  // % S
        if (MODE == 0) {
          ((half_t*)Out)[(((size_t)bb * Hc + hh) * Sc + ss) * DKc + dk] = (half_t)val;
        } else if (MODE == 1) {
          ((half_t*)Out)[(((size_t)bb * Hc + hh) * DKc + dk) * Sc + ss] = (half_t)val;
        } else {
          ((float*)Out)[(size_t)gm * GN + gn] = val;
        }
      }
    }
  }
}

// ---------------------------------------------------------------------------
// Flash attention: one block per (b*H+h, 64-query tile). 128 thr = 4 waves,
// each wave owns 16 query rows. K/V tiles (64 keys) are staged into LDS once
// per block via CDNA5 async global->LDS copies, double-buffered so the next
// tile's DMA overlaps this tile's WMMAs/softmax.
// qh/kh: [BH, S, 64] f16; vt: [BH, 64, S] f16.
// ---------------------------------------------------------------------------
static constexpr int KST = 72;   // K/V LDS row stride in halves (144 B)
static constexpr int PST = 72;   // P   LDS row stride in halves

__device__ __forceinline__ void stage_kv(const half_t* __restrict__ khb,
                                         const half_t* __restrict__ vtb,
                                         half_t* kdst, half_t* vdst,
                                         int kb, int tid) {
  // 64x64 f16 tile each: 512 x b128 ops split over 128 threads (4 each).
#pragma unroll
  for (int j = 0; j < 4; j++) {
    const int idx = j * 128 + tid;        // 0..511
    const int row = idx >> 3;             // 0..63
    const int col = (idx & 7) * 8;        // 0..56 halves
    async_b128(kdst + row * KST + col, khb + (size_t)(kb + row) * DKc + col);
    async_b128(vdst + row * KST + col, vtb + (size_t)row * Sc + kb + col);
  }
}

__global__ __launch_bounds__(128) void flash_attn(const half_t* __restrict__ qh,
                                                  const half_t* __restrict__ kh,
                                                  const half_t* __restrict__ vt,
                                                  half_t* __restrict__ concat) {
  __shared__ __align__(16) half_t Ks[2][64 * KST];
  __shared__ __align__(16) half_t Vs[2][64 * KST];
  __shared__ __align__(16) half_t Plds[4 * 16 * PST];

  const int tid  = threadIdx.x;
  const int lane = tid & 31;
  const int wave = tid >> 5;
  const int l15  = lane & 15;
  const int ah   = lane >> 4;
  const int bh   = blockIdx.y;                 // b*H + h
  const int q0   = blockIdx.x * 64 + wave * 16;

  const half_t* khb = kh + (size_t)bh * Sc * DKc;   // [S][64]
  const half_t* vtb = vt + (size_t)bh * DKc * Sc;   // [64][S]

  // Q fragments for this wave's 16 rows, pre-scaled by 1/sqrt(64)
  const half_t* qrow = qh + ((size_t)bh * Sc + (q0 + l15)) * DKc;
  v16h qf[2];
#pragma unroll
  for (int ds = 0; ds < 2; ds++) {
    Frag16 f;
    f.u[0] = *(const uint4*)(qrow + ds * 32 + ah * 8);
    f.u[1] = *(const uint4*)(qrow + ds * 32 + 16 + ah * 8);
#pragma unroll
    for (int j = 0; j < 16; j++) f.v[j] = f.v[j] * (half_t)0.125f;
    qf[ds] = f.v;
  }

  float mrow[8], lrow[8];
  v8f accO[4];
#pragma unroll
  for (int v = 0; v < 8; v++) { mrow[v] = -3.0e38f; lrow[v] = 0.0f; }
#pragma unroll
  for (int nt = 0; nt < 4; nt++)
#pragma unroll
    for (int v = 0; v < 8; v++) accO[nt][v] = 0.0f;

  half_t* Pw = &Plds[wave * 16 * PST];

  constexpr int NT = Sc / 64;                   // 32 key tiles
  stage_kv(khb, vtb, Ks[0], Vs[0], 0, tid);     // prefetch tile 0

  for (int kt = 0; kt < NT; kt++) {
    const int buf = kt & 1;
    const int kb  = kt * 64;

    // Issue DMA for the next tile, then wait for the current one (async loads
    // complete in order: <=8 outstanding means the older tile's 8 are done).
    if (kt + 1 < NT) {
      stage_kv(khb, vtb, Ks[buf ^ 1], Vs[buf ^ 1], kb + 64, tid);
      wait_async<8>();
    } else {
      wait_async<0>();
    }
    __syncthreads();

    // ---- scores S = (Q*scale) @ K^T : 4 key sub-tiles x 2 K-steps ----
    v8f st[4];
#pragma unroll
    for (int nt = 0; nt < 4; nt++) {
#pragma unroll
      for (int v = 0; v < 8; v++) st[nt][v] = 0.0f;
      const half_t* krow = &Ks[buf][(nt * 16 + l15) * KST];
#pragma unroll
      for (int ds = 0; ds < 2; ds++) {
        Frag16 f;
        f.u[0] = *(const uint4*)(krow + ds * 32 + ah * 8);
        f.u[1] = *(const uint4*)(krow + ds * 32 + ah * 8 + 16);
        st[nt] = WMMA_F16(qf[ds], f.v, st[nt]);
      }
    }

    // ---- online softmax (row stats per VGPR; cols across 16-lane half) ----
    float mnew[8], alpha[8];
#pragma unroll
    for (int v = 0; v < 8; v++) {
      float mx = fmaxf(fmaxf(st[0][v], st[1][v]), fmaxf(st[2][v], st[3][v]));
      mx = fmaxf(mx, __shfl_xor(mx, 1, 32));
      mx = fmaxf(mx, __shfl_xor(mx, 2, 32));
      mx = fmaxf(mx, __shfl_xor(mx, 4, 32));
      mx = fmaxf(mx, __shfl_xor(mx, 8, 32));
      mnew[v]  = fmaxf(mrow[v], mx);
      alpha[v] = __expf(mrow[v] - mnew[v]);
    }
    float ps[4][8];
#pragma unroll
    for (int v = 0; v < 8; v++) {
      float rs = 0.0f;
#pragma unroll
      for (int nt = 0; nt < 4; nt++) {
        ps[nt][v] = __expf(st[nt][v] - mnew[v]);
        rs += ps[nt][v];
      }
      rs += __shfl_xor(rs, 1, 32);
      rs += __shfl_xor(rs, 2, 32);
      rs += __shfl_xor(rs, 4, 32);
      rs += __shfl_xor(rs, 8, 32);
      lrow[v] = lrow[v] * alpha[v] + rs;
      mrow[v] = mnew[v];
#pragma unroll
      for (int nt = 0; nt < 4; nt++) accO[nt][v] *= alpha[v];
    }

    // ---- P: C-layout -> LDS -> A-layout ----
#pragma unroll
    for (int nt = 0; nt < 4; nt++)
#pragma unroll
      for (int v = 0; v < 8; v++)
        Pw[(ah * 8 + v) * PST + nt * 16 + l15] = (half_t)ps[nt][v];
    __syncthreads();

    // ---- O += P @ V (V tile in LDS: contiguous keys per d row) ----
#pragma unroll
    for (int ks = 0; ks < 2; ks++) {
      Frag16 pf;
      pf.u[0] = *(const uint4*)(&Pw[l15 * PST + ks * 32 + ah * 8]);
      pf.u[1] = *(const uint4*)(&Pw[l15 * PST + ks * 32 + 16 + ah * 8]);
#pragma unroll
      for (int nt = 0; nt < 4; nt++) {
        const half_t* vrow = &Vs[buf][(nt * 16 + l15) * KST + ks * 32 + ah * 8];
        Frag16 vf;
        vf.u[0] = *(const uint4*)(vrow);
        vf.u[1] = *(const uint4*)(vrow + 16);
        accO[nt] = WMMA_F16(pf.v, vf.v, accO[nt]);
      }
    }
    __syncthreads();   // P + K/V buffer reuse fence for next iteration
  }

  // ---- normalize and store to concat [B, S, D] f16 ----
  const int bb = bh >> 4, hh = bh & 15;
#pragma unroll
  for (int v = 0; v < 8; v++) {
    const float inv = 1.0f / lrow[v];
    const int ss = q0 + ah * 8 + v;
#pragma unroll
    for (int nt = 0; nt < 4; nt++) {
      const int d = nt * 16 + l15;
      concat[((size_t)bb * Sc + ss) * Dc + hh * DKc + d] = (half_t)(accO[nt][v] * inv);
    }
  }
}

// ---------------------------------------------------------------------------
// Host launcher
// ---------------------------------------------------------------------------
extern "C" void kernel_launch(void* const* d_in, const int* in_sizes, int n_in,
                              void* d_out, int out_size, void* d_ws, size_t ws_size,
                              hipStream_t stream) {
  (void)in_sizes; (void)n_in; (void)out_size; (void)ws_size;
  const float* q  = (const float*)d_in[0];
  const float* k  = (const float*)d_in[1];
  const float* v  = (const float*)d_in[2];
  const float* Wq = (const float*)d_in[3];
  const float* bq = (const float*)d_in[4];
  const float* Wk = (const float*)d_in[5];
  const float* bk = (const float*)d_in[6];
  const float* Wv = (const float*)d_in[7];
  const float* bv = (const float*)d_in[8];
  const float* Wo = (const float*)d_in[9];
  const float* bo = (const float*)d_in[10];

  char* ws = (char*)d_ws;
  const size_t HEAD_BYTES = (size_t)Bc * Hc * Sc * DKc * sizeof(half_t);  // 16 MiB
  half_t* qhd = (half_t*)(ws);
  half_t* khd = (half_t*)(ws + HEAD_BYTES);
  half_t* vtd = (half_t*)(ws + 2 * HEAD_BYTES);
  half_t* ccd = (half_t*)(ws + 3 * HEAD_BYTES);

  dim3 gg(GN / BN, GM / BM);  // (8, 64)
  proj_gemm<0><<<gg, 256, 0, stream>>>(q, Wq, bq, qhd);
  proj_gemm<0><<<gg, 256, 0, stream>>>(k, Wk, bk, khd);
  proj_gemm<1><<<gg, 256, 0, stream>>>(v, Wv, bv, vtd);
  flash_attn<<<dim3(Sc / 64, Bc * Hc), 128, 0, stream>>>(qhd, khd, vtd, ccd);
  proj_gemm<2><<<gg, 256, 0, stream>>>(ccd, Wo, bo, (float*)d_out);
}
